// KANWithFourierAndMLA_29996051595738
// MI455X (gfx1250) — matrix operand order repaired and checked by reading, not compiled, not run on hardware
//
#include <hip/hip_runtime.h>
#include <hip/hip_bf16.h>

// Problem constants (match reference)
#define BB   8192
#define SS   128
#define DDIM 64
#define FFRQ 8
#define HH   2048
#define EE   1088      // D*(2F+1)
#define LLAY 4
#define KKAN 8
#define LDIM 1024
#define OO   512
#define LN_EPS 1e-5f

typedef __attribute__((ext_vector_type(16))) __bf16 v16bf;
typedef __attribute__((ext_vector_type(8)))  __bf16 v8bf;
typedef __attribute__((ext_vector_type(8)))  float  v8f;

// ---------------------------------------------------------------------------
// 1) seq-mean over S + Fourier features, emitted as bf16 [B, 1088]
//    feat = [xm(64) | sin(xm*freqs) (64*8) | cos(xm*freqs) (64*8)]
// ---------------------------------------------------------------------------
__global__ void mean_feat_kernel(const float* __restrict__ x,
                                 const float* __restrict__ freqs,
                                 __bf16* __restrict__ feat) {
    int b = blockIdx.x;
    int d = threadIdx.x;                       // 64 threads: one per D column
    const float* xp = x + (size_t)b * SS * DDIM + d;
    float s = 0.f;
    #pragma unroll 8
    for (int t = 0; t < SS; ++t) s += xp[(size_t)t * DDIM];
    float xm = s * (1.0f / SS);
    __bf16* fr = feat + (size_t)b * EE;
    fr[d] = (__bf16)xm;
    #pragma unroll
    for (int f = 0; f < FFRQ; ++f) {
        float ang = xm * freqs[f];
        fr[DDIM + d * FFRQ + f]               = (__bf16)sinf(ang);
        fr[DDIM + DDIM * FFRQ + d * FFRQ + f] = (__bf16)cosf(ang);
    }
}

// ---------------------------------------------------------------------------
// 2) flat f32 -> bf16 conversion (weights)
// ---------------------------------------------------------------------------
__global__ void f32_to_bf16_kernel(const float* __restrict__ src,
                                   __bf16* __restrict__ dst, size_t n) {
    size_t i = (size_t)blockIdx.x * blockDim.x + threadIdx.x;
    if (i < n) dst[i] = (__bf16)src[i];
}

// ---------------------------------------------------------------------------
// 3) KAN sin-basis: kan[b,h] = sum_k kw[h,k]*sin(ka[h,k]*hval + kb2[h,k]) + kb[h]
// ---------------------------------------------------------------------------
__global__ void kan_kernel(const float* __restrict__ h,
                           const float* __restrict__ kw,
                           const float* __restrict__ kb,
                           const float* __restrict__ ka,
                           const float* __restrict__ kb2,
                           __bf16* __restrict__ kan) {
    int i = blockIdx.x * blockDim.x + threadIdx.x;   // over B*H
    int hc = i & (HH - 1);
    float hv = h[i];
    float acc = kb[hc];
    #pragma unroll
    for (int k = 0; k < KKAN; ++k)
        acc += kw[hc * KKAN + k] * sinf(ka[hc * KKAN + k] * hv + kb2[hc * KKAN + k]);
    kan[i] = (__bf16)acc;
}

// ---------------------------------------------------------------------------
// 4) residual add + LayerNorm over H=2048; writes f32 h (in place) + bf16 copy
// ---------------------------------------------------------------------------
__global__ __launch_bounds__(256) void add_ln_kernel(float* __restrict__ h,
                                                     const float* __restrict__ po,
                                                     const float* __restrict__ g,
                                                     const float* __restrict__ be,
                                                     __bf16* __restrict__ hbf) {
    int row = blockIdx.x;
    int t = threadIdx.x;                      // 256 threads, 8 cols each
    size_t base = (size_t)row * HH;
    float z[8];
    float s = 0.f, s2 = 0.f;
    #pragma unroll
    for (int j = 0; j < 8; ++j) {
        int c = t + j * 256;
        float v = h[base + c] + po[base + c];
        z[j] = v; s += v; s2 += v * v;
    }
    __shared__ float red0[256];
    __shared__ float red1[256];
    red0[t] = s; red1[t] = s2;
    __syncthreads();
    for (int off = 128; off > 0; off >>= 1) {
        if (t < off) { red0[t] += red0[t + off]; red1[t] += red1[t + off]; }
        __syncthreads();
    }
    float mean = red0[0] * (1.f / HH);
    float var  = red1[0] * (1.f / HH) - mean * mean;
    float rstd = rsqrtf(var + LN_EPS);
    #pragma unroll
    for (int j = 0; j < 8; ++j) {
        int c = t + j * 256;
        float o = (z[j] - mean) * rstd * g[c] + be[c];
        h[base + c]   = o;
        hbf[base + c] = (__bf16)o;
    }
}

// ---------------------------------------------------------------------------
// 5) bf16 WMMA GEMM: C[M,N] = A[M,K] * W[N,K]^T + bias[N]
//    Block: 256 threads = 8 waves; block tile 128x128; wave tile 32x64
//    (2 x 4 v_wmma_f32_16x16x32_bf16 accumulators per wave).
//    A tile is streamed into double-buffered LDS with CDNA5 async DMA
//    (global_load_async_to_lds_b128, fenced with s_wait_asynccnt), so the
//    next tile's fetch overlaps this tile's WMMAs. B fragments are read
//    directly from global (weights are L2 resident).
// ---------------------------------------------------------------------------
#define TILE_HALVES (128 * 32)          // one A tile: 128 rows x 32 K (bf16)

template<bool OUT_BF16>
__global__ __launch_bounds__(256) void gemm_wmma(const __bf16* __restrict__ A,
                                                 const __bf16* __restrict__ W,
                                                 const float* __restrict__ bias,
                                                 void* __restrict__ Cv,
                                                 int N, int K) {
    __shared__ __align__(32) __bf16 As[2][TILE_HALVES];   // 2 x 8 KB

    const int tid   = threadIdx.x;
    const int lane  = tid & 31;
    const int wave  = tid >> 5;        // 0..7
    const int wm    = wave >> 1;       // 0..3  (M direction)
    const int wn    = wave & 1;        // 0..1  (N direction)
    const int lmod  = lane & 15;
    const int lhalf = lane >> 4;       // 0/1: half-wave select

    const int blockM = blockIdx.x * 128;
    const int blockN = blockIdx.y * 128;
    const int n0 = blockN + wn * 64;

    v8f acc[2][4];
    #pragma unroll
    for (int mi = 0; mi < 2; ++mi)
        #pragma unroll
        for (int ni = 0; ni < 4; ++ni)
            acc[mi][ni] = (v8f){0.f,0.f,0.f,0.f,0.f,0.f,0.f,0.f};

    // A staging: thread t owns 32 contiguous bytes of the tile:
    // row = t/2, halves (t&1)*16 .. +15  -> LDS byte offset t*32
    const int ldRow = tid >> 1;
    const int ldCol = (tid & 1) * 16;
    const __bf16* aG = A + (size_t)(blockM + ldRow) * K + ldCol;

    // 32-bit LDS byte addresses for async DMA (low 32 bits of flat LDS ptr)
    const unsigned ldsDst0 = (unsigned)(size_t)(const void*)&As[0][0] + (unsigned)tid * 32u;
    const unsigned ldsDst1 = ldsDst0 + (unsigned)(TILE_HALVES * 2);

    const int nsteps = K >> 5;

    // Prologue: async-stage tile 0 (2 x 16B per thread)
    {
        asm volatile("global_load_async_to_lds_b128 %0, %1, off"
                     :: "v"(ldsDst0), "v"(aG) : "memory");
        asm volatile("global_load_async_to_lds_b128 %0, %1, off"
                     :: "v"(ldsDst0 + 16u), "v"(aG + 8) : "memory");
    }

    for (int kt = 0; kt < nsteps; ++kt) {
        const int buf = kt & 1;
        // Prefetch next tile into the other buffer, then wait for current
        if (kt + 1 < nsteps) {
            const unsigned dst = buf ? ldsDst0 : ldsDst1;
            const __bf16* g = aG + (size_t)(kt + 1) * 32;
            asm volatile("global_load_async_to_lds_b128 %0, %1, off"
                         :: "v"(dst), "v"(g) : "memory");
            asm volatile("global_load_async_to_lds_b128 %0, %1, off"
                         :: "v"(dst + 16u), "v"(g + 8) : "memory");
            asm volatile("s_wait_asynccnt 0x2" ::: "memory");  // current tile done
        } else {
            asm volatile("s_wait_asynccnt 0x0" ::: "memory");
        }
        __syncthreads();   // current tile visible to all waves

        // B fragments: N = lane&15, K = lhalf*16 + i (contiguous, matches A)
        v16bf bfrag[4];
        #pragma unroll
        for (int ni = 0; ni < 4; ++ni) {
            int col = n0 + ni * 16 + lmod;
            bfrag[ni] = *(const v16bf*)(W + (size_t)col * K + kt * 32 + lhalf * 16);
        }

        // A fragments from LDS: M = lane&15,
        // K = lhalf*16 + i (contiguous, matches B)
        union { v16bf v; v8bf h8[2]; } afrag[2];
        #pragma unroll
        for (int mi = 0; mi < 2; ++mi) {
            int r = wm * 32 + mi * 16 + lmod;
            const __bf16* ap = &As[buf][r * 32 + lhalf * 16];
            afrag[mi].h8[0] = *(const v8bf*)(ap);
            afrag[mi].h8[1] = *(const v8bf*)(ap + 8);
        }

        #pragma unroll
        for (int mi = 0; mi < 2; ++mi)
            #pragma unroll
            for (int ni = 0; ni < 4; ++ni)
                acc[mi][ni] = __builtin_amdgcn_wmma_f32_16x16x32_bf16(
                    false, afrag[mi].v, false, bfrag[ni],
                    (short)0, acc[mi][ni], false, false);

        __syncthreads();   // reads done before next iter's async overwrites buf
    }

    // Epilogue: C/D layout — N = lane&15; M = r + 8*(lane>=16)
    const int m0 = blockM + wm * 32;
    #pragma unroll
    for (int ni = 0; ni < 4; ++ni) {
        int col = n0 + ni * 16 + lmod;
        float bv = bias[col];
        #pragma unroll
        for (int mi = 0; mi < 2; ++mi) {
            int rbase = m0 + mi * 16 + lhalf * 8;
            #pragma unroll
            for (int r = 0; r < 8; ++r) {
                float v = acc[mi][ni][r] + bv;
                size_t idx = (size_t)(rbase + r) * N + col;
                if (OUT_BF16) ((__bf16*)Cv)[idx] = (__bf16)v;
                else          ((float*)Cv)[idx]  = v;
            }
        }
    }
}

// ---------------------------------------------------------------------------
// Launch
// ---------------------------------------------------------------------------
extern "C" void kernel_launch(void* const* d_in, const int* in_sizes, int n_in,
                              void* d_out, int out_size, void* d_ws, size_t ws_size,
                              hipStream_t stream) {
    const float* x        = (const float*)d_in[0];
    const float* freqs    = (const float*)d_in[1];
    const float* proj_w   = (const float*)d_in[2];
    const float* proj_b   = (const float*)d_in[3];
    const float* kan_w    = (const float*)d_in[4];
    const float* kan_bias = (const float*)d_in[5];
    const float* kan_a    = (const float*)d_in[6];
    const float* kan_b2   = (const float*)d_in[7];
    // d_in[8..11] = q_w,q_b,k_w,k_b : dead (softmax over singleton axis == 1)
    const float* v_w      = (const float*)d_in[12];
    const float* v_b      = (const float*)d_in[13];
    const float* in_w     = (const float*)d_in[14];
    const float* in_b     = (const float*)d_in[15];
    const float* mo_w     = (const float*)d_in[16];
    const float* mo_b     = (const float*)d_in[17];
    const float* out_w    = (const float*)d_in[18];
    const float* out_b    = (const float*)d_in[19];
    const float* ln_g     = (const float*)d_in[20];
    const float* ln_be    = (const float*)d_in[21];
    const float* fc_w     = (const float*)d_in[22];
    const float* fc_b     = (const float*)d_in[23];
    float* out = (float*)d_out;

    char* ws = (char*)d_ws;
    size_t off = 0;
    auto carve = [&](size_t bytes) -> char* {
        char* p = ws + off;
        off += bytes;
        off = (off + 255) & ~(size_t)255;
        return p;
    };

    __bf16* featbf = (__bf16*)carve((size_t)BB * EE * 2);
    float*  hbuf   = (float*) carve((size_t)BB * HH * 4);
    __bf16* hbf    = (__bf16*)carve((size_t)BB * HH * 2);
    __bf16* kanbf  = (__bf16*)carve((size_t)BB * HH * 2);
    __bf16* vbf    = (__bf16*)carve((size_t)BB * LDIM * 2);
    __bf16* vibf   = (__bf16*)carve((size_t)BB * LDIM * 2);
    __bf16* mobf   = (__bf16*)carve((size_t)BB * LDIM * 2);
    float*  pobuf  = (float*) carve((size_t)BB * HH * 4);
    __bf16* wproj  = (__bf16*)carve((size_t)HH * EE * 2);
    __bf16* wv     = (__bf16*)carve((size_t)LLAY * LDIM * HH * 2);
    __bf16* wvi    = (__bf16*)carve((size_t)LLAY * LDIM * LDIM * 2);
    __bf16* wmo    = (__bf16*)carve((size_t)LLAY * LDIM * LDIM * 2);
    __bf16* wout   = (__bf16*)carve((size_t)LLAY * HH * LDIM * 2);
    __bf16* wfc    = (__bf16*)carve((size_t)OO * HH * 2);
    (void)ws_size; (void)in_sizes; (void)n_in; (void)out_size;

    auto conv = [&](const float* src, __bf16* dst, size_t n) {
        f32_to_bf16_kernel<<<dim3((unsigned)((n + 255) / 256)), 256, 0, stream>>>(src, dst, n);
    };
    // Weight conversions (f32 -> bf16, keep [N,K] layout)
    conv(proj_w, wproj, (size_t)HH * EE);
    conv(v_w,    wv,    (size_t)LLAY * LDIM * HH);
    for (int l = 0; l < LLAY; ++l)   // only the V slice of in_proj is live
        conv(in_w + (size_t)l * 3 * LDIM * LDIM + (size_t)2 * LDIM * LDIM,
             wvi + (size_t)l * LDIM * LDIM, (size_t)LDIM * LDIM);
    conv(mo_w,   wmo,   (size_t)LLAY * LDIM * LDIM);
    conv(out_w,  wout,  (size_t)LLAY * HH * LDIM);
    conv(fc_w,   wfc,   (size_t)OO * HH);

    // Features
    mean_feat_kernel<<<dim3(BB), 64, 0, stream>>>(x, freqs, featbf);

    auto gemm_f = [&](const __bf16* A, const __bf16* Wt, const float* bias,
                      float* C, int N, int K) {
        gemm_wmma<false><<<dim3(BB / 128, N / 128), 256, 0, stream>>>(A, Wt, bias, (void*)C, N, K);
    };
    auto gemm_b = [&](const __bf16* A, const __bf16* Wt, const float* bias,
                      __bf16* C, int N, int K) {
        gemm_wmma<true><<<dim3(BB / 128, N / 128), 256, 0, stream>>>(A, Wt, bias, (void*)C, N, K);
    };

    // h = feat @ proj_w^T + proj_b
    gemm_f(featbf, wproj, proj_b, hbuf, HH, EE);

    for (int l = 0; l < LLAY; ++l) {
        kan_kernel<<<dim3((BB * HH) / 256), 256, 0, stream>>>(
            hbuf,
            kan_w   + (size_t)l * HH * KKAN,
            kan_bias+ (size_t)l * HH,
            kan_a   + (size_t)l * HH * KKAN,
            kan_b2  + (size_t)l * HH * KKAN,
            kanbf);

        // v = kan @ v_w^T + v_b
        gemm_b(kanbf, wv + (size_t)l * LDIM * HH, v_b + (size_t)l * LDIM,
               vbf, LDIM, HH);
        // vi = v @ in_w[2LD:]^T + in_b[2LD:]   (attention is identity on vi)
        gemm_b(vbf, wvi + (size_t)l * LDIM * LDIM,
               in_b + (size_t)l * 3 * LDIM + 2 * LDIM, vibf, LDIM, LDIM);
        // mo = vi @ mo_w^T + mo_b
        gemm_b(vibf, wmo + (size_t)l * LDIM * LDIM, mo_b + (size_t)l * LDIM,
               mobf, LDIM, LDIM);
        // po = mo @ out_w^T + out_b
        gemm_f(mobf, wout + (size_t)l * HH * LDIM, out_b + (size_t)l * HH,
               pobuf, HH, LDIM);
        // h = LN(h + po); also bf16 copy
        add_ln_kernel<<<dim3(BB), 256, 0, stream>>>(
            hbuf, pobuf, ln_g + (size_t)l * HH, ln_be + (size_t)l * HH, hbf);
    }

    // out = h @ fc_w^T + fc_b
    gemm_f(hbf, wfc, fc_b, out, OO, HH);
}
